// W_Attention_12429635355151
// MI455X (gfx1250) — matrix operand order repaired and checked by reading, not compiled, not run on hardware
//
#include <hip/hip_runtime.h>
#include <hip/hip_bf16.h>
#include <stdint.h>

typedef __bf16 bf16_t;
typedef bf16_t v16bf __attribute__((ext_vector_type(16)));
typedef bf16_t v8bf  __attribute__((ext_vector_type(8)));
typedef float  v8f   __attribute__((ext_vector_type(8)));

#define DEV static __device__ __forceinline__

constexpr int Bn = 8, Cn = 256, Hn = 128, Wn = 256;   // x: [B,C,H,W]
constexpr int XS_STR = 264, VS_STR = 264, AT_STR = 264, QS_STR = 32;
constexpr int XS_OFF = 0;
constexpr int VS_OFF = XS_OFF + Wn * XS_STR * 2;      // xs: [pixel][chan] bf16
constexpr int QS_OFF = VS_OFF + Cn * VS_STR * 2;      // vs: [c][j] bf16
constexpr int KS_OFF = QS_OFF + Wn * QS_STR * 2;      // qs: [pixel][32] bf16
constexpr int LDS_BYTES = KS_OFF + Wn * QS_STR * 2;   // 303104 bytes (<320KB/WG)

// Native converts: let the backend emit v_cvt_pk_bf16_f32-class instructions
// instead of a 4-VALU-op RNE emulation per element.
DEV bf16_t bf(float f) { return (bf16_t)f; }
DEV v8bf cvt8(v8f v)   { return __builtin_convertvector(v, v8bf); }

DEV v8f wmma_bf16(v16bf a, v16bf b, v8f c) {
  return __builtin_amdgcn_wmma_f32_16x16x32_bf16(false, a, false, b,
                                                 (short)0, c, false, false);
}

// A-fragment 16x32 bf16 from LDS, storage row-major [m][k].
// lanes 0-15 hold K chunks {k0..k0+7, k0+16..k0+23}; lanes 16-31 the +8 chunks.
DEV v16bf lds_a(const bf16_t* base, int stride, int row, int k0, int half) {
  const v8bf lo = *(const v8bf*)(base + row * stride + k0 + half * 8);
  const v8bf hi = *(const v8bf*)(base + row * stride + k0 + 16 + half * 8);
  v16bf a;
  #pragma unroll
  for (int i = 0; i < 8; ++i) { a[i] = lo[i]; a[i + 8] = hi[i]; }
  return a;
}

// A-fragment from global fp32 weights, row-major [m][256], packed-convert to bf16.
DEV v16bf glb_a_f32(const float* base, int row, int k0, int half) {
  const float* p = base + row * 256 + k0 + half * 8;   // 32B aligned
  const v8bf lo = cvt8(*(const v8f*)(p));
  const v8bf hi = cvt8(*(const v8f*)(p + 16));
  v16bf a;
  #pragma unroll
  for (int i = 0; i < 8; ++i) { a[i] = lo[i]; a[i + 8] = hi[i]; }
  return a;
}

// B-fragment 32x16 bf16 from LDS, storage [n][k] with k contiguous.
// lanes 0-15 hold K chunks {k0..k0+7, k0+16..k0+23}; lanes 16-31 the +8 chunks.
DEV v16bf lds_b(const bf16_t* base, int stride, int col, int k0, int half) {
  const bf16_t* p = base + col * stride + k0 + half * 8;
  const v8bf lo = *(const v8bf*)(p);
  const v8bf hi = *(const v8bf*)(p + 16);
  v16bf b;
  #pragma unroll
  for (int i = 0; i < 8; ++i) { b[i] = lo[i]; b[i + 8] = hi[i]; }
  return b;
}

__global__ __launch_bounds__(256)
void W_Attention_kernel(const float* __restrict__ x,
                        const float* __restrict__ Wq, const float* __restrict__ bq,
                        const float* __restrict__ Wk, const float* __restrict__ bk,
                        const float* __restrict__ Wv, const float* __restrict__ bv,
                        const float* __restrict__ gamma,
                        float* __restrict__ out) {
  extern __shared__ char smem[];
  bf16_t* xs = (bf16_t*)(smem + XS_OFF);
  bf16_t* vs = (bf16_t*)(smem + VS_OFF);
  bf16_t* qs = (bf16_t*)(smem + QS_OFF);
  bf16_t* ks = (bf16_t*)(smem + KS_OFF);
  bf16_t* at = xs;  // attn reuses xs space after v-projection (128 rows/pass)

  const int tid  = threadIdx.x;
  const int wave = tid >> 5;
  const int lane = tid & 31;
  const int half = lane >> 4;
  const int l    = lane & 15;

  const int bh = blockIdx.x;
  const int b  = bh >> 7;      // H = 128
  const int h  = bh & 127;
  const float* xg = x   + (size_t)b * Cn * Hn * Wn + (size_t)h * Wn;
  float*       og = out + (size_t)b * Cn * Hn * Wn + (size_t)h * Wn;
  const float  g  = gamma[0];

  // ---- Stage 1: x row-slab [C=256][W=256] -> LDS bf16 as xs[pixel][chan] ----
  #pragma unroll 4
  for (int q = 0; q < 64; ++q) {
    int f = q * 256 + tid;          // float4 index over 65536 elements
    int c = f >> 6;                 // (f*4) >> 8
    int p = (f & 63) << 2;
    const float4 d = *(const float4*)(xg + (size_t)c * Hn * Wn + p);
    xs[(p + 0) * XS_STR + c] = bf(d.x);
    xs[(p + 1) * XS_STR + c] = bf(d.y);
    xs[(p + 2) * XS_STR + c] = bf(d.z);
    xs[(p + 3) * XS_STR + c] = bf(d.w);
  }
  __syncthreads();

  // ---- Stage 2: q = Wq@x + bq, k = Wk@x + bk  ->  qs/ks [pixel][32] ----
  {
    const bool isk = (wave & 1) != 0;
    const float* Wqk = isk ? Wk : Wq;
    const float* bqk = isk ? bk : bq;
    bf16_t* dst = isk ? ks : qs;
    const int mt = (wave >> 1) & 1;
    const int nh = (wave >> 2) & 1;

    v16bf aW[8];
    #pragma unroll
    for (int kc = 0; kc < 8; ++kc) aW[kc] = glb_a_f32(Wqk, mt * 16 + l, kc * 32, half);

    for (int t = 0; t < 8; ++t) {
      const int n0 = (nh * 8 + t) * 16;
      v8f acc;
      #pragma unroll
      for (int r = 0; r < 8; ++r) acc[r] = bqk[mt * 16 + half * 8 + r];
      #pragma unroll
      for (int kc = 0; kc < 8; ++kc)
        acc = wmma_bf16(aW[kc], lds_b(xs, XS_STR, n0 + l, kc * 32, half), acc);
      *(v8bf*)(dst + (n0 + l) * QS_STR + mt * 16 + half * 8) = cvt8(acc);
    }
  }

  // ---- Stage 3: v = Wv@x + bv -> vs[c][j] (transposed scatter store) ----
  for (int mi = 0; mi < 2; ++mi) {
    const int mt = wave + mi * 8;
    v16bf aV[8];
    #pragma unroll
    for (int kc = 0; kc < 8; ++kc) aV[kc] = glb_a_f32(Wv, mt * 16 + l, kc * 32, half);

    for (int nt = 0; nt < 16; ++nt) {
      v8f acc;
      #pragma unroll
      for (int r = 0; r < 8; ++r) acc[r] = bv[mt * 16 + half * 8 + r];
      #pragma unroll
      for (int kc = 0; kc < 8; ++kc)
        acc = wmma_bf16(aV[kc], lds_b(xs, XS_STR, nt * 16 + l, kc * 32, half), acc);
      const v8bf o = cvt8(acc);
      #pragma unroll
      for (int r = 0; r < 8; ++r)
        vs[(mt * 16 + half * 8 + r) * VS_STR + nt * 16 + l] = o[r];
    }
  }
  __syncthreads();  // xs is now dead; qs/ks/vs ready

  // ---- Stages 4+5: energy -> softmax -> out, 128 query pixels per pass ----
  #pragma unroll 1
  for (int pass = 0; pass < 2; ++pass) {
    const int i0 = pass * 128 + wave * 16;   // this wave's i-tile (global)

    // energy E[i0..i0+15][*] = q^T k, K = 32 (one WMMA per j-tile)
    const v16bf aQ = lds_a(qs, QS_STR, i0 + l, 0, half);
    v8f E[16];
    #pragma unroll
    for (int jt = 0; jt < 16; ++jt) {
      v8f z = {0.f, 0.f, 0.f, 0.f, 0.f, 0.f, 0.f, 0.f};
      E[jt] = wmma_bf16(aQ, lds_b(ks, QS_STR, jt * 16 + l, 0, half), z);
    }

    // row-wise softmax: row i lives at (vgpr r, lane-half), 16 lanes wide
    float inv[8];
    #pragma unroll
    for (int r = 0; r < 8; ++r) {
      float m = E[0][r];
      #pragma unroll
      for (int jt = 1; jt < 16; ++jt) m = fmaxf(m, E[jt][r]);
      m = fmaxf(m, __shfl_xor(m, 1)); m = fmaxf(m, __shfl_xor(m, 2));
      m = fmaxf(m, __shfl_xor(m, 4)); m = fmaxf(m, __shfl_xor(m, 8));
      float s = 0.f;
      #pragma unroll
      for (int jt = 0; jt < 16; ++jt) {
        float e = __expf(E[jt][r] - m);
        E[jt][r] = e;
        s += e;
      }
      s += __shfl_xor(s, 1); s += __shfl_xor(s, 2);
      s += __shfl_xor(s, 4); s += __shfl_xor(s, 8);
      inv[r] = 1.f / s;
    }

    __syncthreads();  // previous pass's attn readers are done
    #pragma unroll
    for (int jt = 0; jt < 16; ++jt)
      #pragma unroll
      for (int r = 0; r < 8; ++r)
        at[(wave * 16 + half * 8 + r) * AT_STR + jt * 16 + l] = bf(E[jt][r] * inv[r]);
    __syncthreads();

    // out^T[i][c] = attn(A, [i][j]) x V^T(B from vs[c][j]), K = 256
    v16bf aA[8];
    #pragma unroll
    for (int kc = 0; kc < 8; ++kc)
      aA[kc] = lds_a(at, AT_STR, wave * 16 + l, kc * 32, half);

    for (int ct = 0; ct < 16; ++ct) {
      v8f acc = {0.f, 0.f, 0.f, 0.f, 0.f, 0.f, 0.f, 0.f};
      #pragma unroll
      for (int kc = 0; kc < 8; ++kc)
        acc = wmma_bf16(aA[kc], lds_b(vs, VS_STR, ct * 16 + l, kc * 32, half), acc);

      // lane: c fixed = ct*16+l, i = i0 + half*8 + r contiguous -> 2x b128
      const int c = ct * 16 + l;
      const size_t goff = (size_t)c * Hn * Wn + (size_t)(i0 + half * 8);
      const float4 x0 = *(const float4*)(xg + goff);
      const float4 x1 = *(const float4*)(xg + goff + 4);
      float4 o0, o1;
      o0.x = fmaf(g, acc[0], x0.x); o0.y = fmaf(g, acc[1], x0.y);
      o0.z = fmaf(g, acc[2], x0.z); o0.w = fmaf(g, acc[3], x0.w);
      o1.x = fmaf(g, acc[4], x1.x); o1.y = fmaf(g, acc[5], x1.y);
      o1.z = fmaf(g, acc[6], x1.z); o1.w = fmaf(g, acc[7], x1.w);
      *(float4*)(og + goff)     = o0;
      *(float4*)(og + goff + 4) = o1;
    }
  }
}

extern "C" void kernel_launch(void* const* d_in, const int* in_sizes, int n_in,
                              void* d_out, int out_size, void* d_ws, size_t ws_size,
                              hipStream_t stream) {
  const float* x     = (const float*)d_in[0];
  const float* Wq    = (const float*)d_in[1];
  const float* bq    = (const float*)d_in[2];
  const float* Wk    = (const float*)d_in[3];
  const float* bk    = (const float*)d_in[4];
  const float* Wv    = (const float*)d_in[5];
  const float* bv    = (const float*)d_in[6];
  const float* gamma = (const float*)d_in[7];
  float* out = (float*)d_out;

  dim3 grid(Bn * Hn);   // 1024 workgroups, one per (b,h) attention row
  dim3 block(256);      // 8 wave32s
  W_Attention_kernel<<<grid, block, LDS_BYTES, stream>>>(
      x, Wq, bq, Wk, bk, Wv, bv, gamma, out);
}
